// DecoupledAttention_52020643889684
// MI455X (gfx1250) — matrix operand order repaired and checked by reading, not compiled, not run on hardware
//
#include <hip/hip_runtime.h>
#include <math.h>

// ---------------------------------------------------------------------------
// Types for CDNA5 WMMA (gfx1250, wave32)
// ---------------------------------------------------------------------------
typedef __bf16 bf16_t;
typedef __attribute__((ext_vector_type(16))) __bf16 v16bf;
typedef __attribute__((ext_vector_type(8)))  __bf16 v8bf;
typedef __attribute__((ext_vector_type(8)))  float  v8f;

__device__ __forceinline__ v16bf concat8(v8bf a, v8bf b) {
  return __builtin_shufflevector(a, b, 0,1,2,3,4,5,6,7,8,9,10,11,12,13,14,15);
}

__device__ __forceinline__ v8f wmma_bf16(v16bf a, v16bf b, v8f c) {
  // (neg_a, A, neg_b, B, c_mod, C, reuse_a, reuse_b)
  return __builtin_amdgcn_wmma_f32_16x16x32_bf16(false, a, false, b, (short)0, c,
                                                 false, false);
}

// A fragment: rows m0..m0+15, cols k0..k0+31 of a row-major [ld] bf16 matrix.
// A 16x32 layout (contiguous k order, matching B loaders): lanes 0-15 -> K kb..kb+15 with kb=0;
// lanes 16-31 same with kb=16.
__device__ __forceinline__ v16bf load_frag_a(const bf16_t* p, int ld, int m0, int k0) {
  int l = threadIdx.x & 31;
  int row = m0 + (l & 15);
  int kb  = k0 + ((l < 16) ? 0 : 16);
  const bf16_t* r = p + (size_t)row * ld + kb;
  v8bf lo = *(const v8bf*)(r);        // K kb .. kb+7
  v8bf hi = *(const v8bf*)(r + 8);    // K kb+8 .. kb+15
  return concat8(lo, hi);
}

// B fragment for B = W^T (B[k][n] = W[n][k]); W row-major [ld].
// Lane n = n0 + (l&15); K = kb..kb+15 contiguous within row n. kb=(l<16)?0:16.
__device__ __forceinline__ v16bf load_frag_bT(const bf16_t* w, int ld, int n0, int k0) {
  int l = threadIdx.x & 31;
  int row = n0 + (l & 15);
  int kb  = k0 + ((l < 16) ? 0 : 16);
  const bf16_t* r = w + (size_t)row * ld + kb;
  v8bf lo = *(const v8bf*)(r);
  v8bf hi = *(const v8bf*)(r + 8);
  return concat8(lo, hi);
}

// B fragment from an LDS tile (16 rows x 64 cols bf16, row-major, pitch 64).
__device__ __forceinline__ v16bf load_frag_b_lds(const bf16_t* tile, int k0) {
  int l = threadIdx.x & 31;
  int row = l & 15;
  int kb  = k0 + ((l < 16) ? 0 : 16);
  const bf16_t* r = tile + row * 64 + kb;
  v8bf lo = *(const v8bf*)(r);
  v8bf hi = *(const v8bf*)(r + 8);
  return concat8(lo, hi);
}

// ---------------------------------------------------------------------------
// Problem constants
// ---------------------------------------------------------------------------
#define BB 2
#define SS 1024
#define DD 1024
#define HH 16
#define HD 64
#define ROWS (BB * SS)          // 2048
#define RMS_EPS 1.1920929e-07f
#define INV_SQRT_N 0.03125f     // 1/sqrt(1024)
#define ESCALE 0.015625f        // 1/64

// ---------------------------------------------------------------------------
// 1. RMSNorm -> bf16 h_norm ; also zero the zeros_like(V) output section
// ---------------------------------------------------------------------------
__global__ __launch_bounds__(256) void k_rmsnorm(const float* __restrict__ h,
                                                 bf16_t* __restrict__ hnbf,
                                                 float* __restrict__ vzero) {
  int row = blockIdx.x;           // 0..2047
  int t = threadIdx.x;            // 256 threads, 4 elems each
  __shared__ float red[256];
  const float* hr = h + (size_t)row * DD;
  float4 v = ((const float4*)hr)[t];
  red[t] = v.x*v.x + v.y*v.y + v.z*v.z + v.w*v.w;
  __syncthreads();
  for (int o = 128; o > 0; o >>= 1) { if (t < o) red[t] += red[t + o]; __syncthreads(); }
  float inv = rsqrtf(red[0] * (1.0f / DD) + RMS_EPS);
  bf16_t* o = hnbf + (size_t)row * DD + t * 4;
  o[0] = (bf16_t)(v.x * inv); o[1] = (bf16_t)(v.y * inv);
  o[2] = (bf16_t)(v.z * inv); o[3] = (bf16_t)(v.w * inv);
  float4 z = make_float4(0.f, 0.f, 0.f, 0.f);
  ((float4*)(vzero + (size_t)row * DD))[t] = z;
}

// ---------------------------------------------------------------------------
// 2. float -> bf16 convert (weights)
// ---------------------------------------------------------------------------
__global__ __launch_bounds__(256) void k_tobf16(const float* __restrict__ s,
                                                bf16_t* __restrict__ d) {
  size_t i = (size_t)blockIdx.x * blockDim.x + threadIdx.x;
  float4 v = ((const float4*)s)[i];
  bf16_t* o = d + i * 4;
  o[0] = (bf16_t)v.x; o[1] = (bf16_t)v.y; o[2] = (bf16_t)v.z; o[3] = (bf16_t)v.w;
}

// ---------------------------------------------------------------------------
// 3. Transpose h_norm (per batch): hnT[(b*D + d)*S + s] = hn[(b*S+s)*D + d]
// ---------------------------------------------------------------------------
__global__ __launch_bounds__(256) void k_transpose(const bf16_t* __restrict__ hnbf,
                                                   bf16_t* __restrict__ hnT) {
  int orow = blockIdx.x;                 // b*1024 + d
  int b = orow >> 10, d = orow & 1023;
  for (int i = threadIdx.x; i < SS; i += 256)
    hnT[(size_t)orow * SS + i] = hnbf[((size_t)(b * SS + i)) * DD + d];
}

// ---------------------------------------------------------------------------
// 4. Projection GEMM: P = hn @ W^T and Pr = hn @ Wc^T ; emit bf16 (scaled)
//    and f32 (P - Pr) to d_out with split-heads indexing. One wave = 16x16 tile.
// ---------------------------------------------------------------------------
__global__ __launch_bounds__(256) void k_proj(const bf16_t* __restrict__ A,
                                              const bf16_t* __restrict__ W,
                                              const bf16_t* __restrict__ Wc,
                                              bf16_t* __restrict__ Pbf,
                                              bf16_t* __restrict__ Prbf,
                                              float* __restrict__ outdiff) {
  int wid = (int)((blockIdx.x * blockDim.x + threadIdx.x) >> 5);
  int tm = wid >> 6;      // 128 M-tiles
  int tn = wid & 63;      // 64  N-tiles
  int m0 = tm * 16, n0 = tn * 16;
  v8f acc = {}; v8f accr = {};
  for (int k0 = 0; k0 < DD; k0 += 32) {
    v16bf a  = load_frag_a (A, DD, m0, k0);
    v16bf b  = load_frag_bT(W, DD, n0, k0);
    v16bf bc = load_frag_bT(Wc, DD, n0, k0);
    acc  = wmma_bf16(a, b,  acc);
    accr = wmma_bf16(a, bc, accr);
  }
  int l = threadIdx.x & 31;
  int n  = n0 + (l & 15);
  int mb = m0 + ((l < 16) ? 0 : 8);
  int head = n >> 6, hd = n & 63;
  for (int v = 0; v < 8; ++v) {
    int row = mb + v;
    int bb = row >> 10, s = row & 1023;
    float p  = INV_SQRT_N * acc[v];
    float pr = INV_SQRT_N * accr[v];
    Pbf [(size_t)row * DD + n] = (bf16_t)p;
    Prbf[(size_t)row * DD + n] = (bf16_t)pr;
    outdiff[(((size_t)(bb * HH + head)) * SS + s) * HD + hd] = p - pr;
  }
}

// ---------------------------------------------------------------------------
// 5. Energy: per wave one (bh, qtile, ktile). K/Kr tiles staged to LDS via
//    CDNA5 async global->LDS loads. Writes logits_diff (output) and
//    scaling*energy (scratch in att_minus region).
// ---------------------------------------------------------------------------
__global__ __launch_bounds__(256) void k_energy(const bf16_t* __restrict__ Qbf,
                                                const bf16_t* __restrict__ Kbf,
                                                const bf16_t* __restrict__ Qrbf,
                                                const bf16_t* __restrict__ Krbf,
                                                float* __restrict__ ldiff,
                                                float* __restrict__ logits) {
  __shared__ __align__(16) bf16_t smem[8 * 2 * 16 * 64];   // 32 KB: 8 waves x (K,Kr) tiles
  int blk = blockIdx.x;
  int ktg = blk & 7;
  int qt  = (blk >> 3) & 63;
  int bh  = blk >> 9;                      // 0..31
  int w   = threadIdx.x >> 5;
  int l   = threadIdx.x & 31;
  int kt  = ktg * 8 + w;
  int b = bh >> 4, h = bh & 15;
  int q0 = qt * 16, k0 = kt * 16;

  bf16_t* tK  = &smem[w * 2048];
  bf16_t* tKr = tK + 1024;
  unsigned ldsK  = (unsigned)(size_t)tK  + (unsigned)((l >> 3) * 128 + (l & 7) * 16);
  unsigned ldsKr = ldsK + 2048u;
  const bf16_t* gK  = Kbf  + ((size_t)(b * SS + k0 + (l >> 3))) * DD + h * HD + (l & 7) * 8;
  const bf16_t* gKr = Krbf + ((size_t)(b * SS + k0 + (l >> 3))) * DD + h * HD + (l & 7) * 8;
  #pragma unroll
  for (int i = 0; i < 4; ++i) {
    unsigned long long aK  = (unsigned long long)(size_t)(gK  + (size_t)i * 4 * DD);
    unsigned long long aKr = (unsigned long long)(size_t)(gKr + (size_t)i * 4 * DD);
    asm volatile("global_load_async_to_lds_b128 %0, %1, off"
                 :: "v"(ldsK + (unsigned)(i * 512)), "v"(aK) : "memory");
    asm volatile("global_load_async_to_lds_b128 %0, %1, off"
                 :: "v"(ldsKr + (unsigned)(i * 512)), "v"(aKr) : "memory");
  }
  asm volatile("s_wait_asynccnt 0" ::: "memory");

  v8f e = {}, er = {};
  for (int kk = 0; kk < HD; kk += 32) {
    v16bf aq  = load_frag_a(Qbf,  DD, b * SS + q0, h * HD + kk);
    v16bf aqr = load_frag_a(Qrbf, DD, b * SS + q0, h * HD + kk);
    v16bf bk  = load_frag_b_lds(tK,  kk);
    v16bf bkr = load_frag_b_lds(tKr, kk);
    e  = wmma_bf16(aq,  bk,  e);
    er = wmma_bf16(aqr, bkr, er);
  }
  int n  = k0 + (l & 15);
  int mb = (l < 16) ? 0 : 8;
  size_t obase = (size_t)bh * SS * SS;
  for (int v = 0; v < 8; ++v) {
    int q = q0 + mb + v;
    logits[obase + (size_t)q * SS + n] = ESCALE * e[v];
    ldiff [obase + (size_t)q * SS + n] = ESCALE * (e[v] - er[v]);
  }
}

// ---------------------------------------------------------------------------
// 6. Per-row (max, sum exp) of logits (for attn softmax in attn@V)
// ---------------------------------------------------------------------------
__global__ __launch_bounds__(256) void k_rowstats(const float* __restrict__ logits,
                                                  float2* __restrict__ stats) {
  int row = blockIdx.x; int t = threadIdx.x;
  __shared__ float red[256];
  float4 v = ((const float4*)(logits + (size_t)row * SS))[t];
  red[t] = fmaxf(fmaxf(v.x, v.y), fmaxf(v.z, v.w));
  __syncthreads();
  for (int o = 128; o > 0; o >>= 1) { if (t < o) red[t] = fmaxf(red[t], red[t+o]); __syncthreads(); }
  float m = red[0];
  __syncthreads();
  red[t] = __expf(v.x - m) + __expf(v.y - m) + __expf(v.z - m) + __expf(v.w - m);
  __syncthreads();
  for (int o = 128; o > 0; o >>= 1) { if (t < o) red[t] += red[t+o]; __syncthreads(); }
  if (t == 0) stats[row] = make_float2(m, red[0]);
}

// ---------------------------------------------------------------------------
// 7. Row softmax: att_minus = softmax(logits_diff)
// ---------------------------------------------------------------------------
__global__ __launch_bounds__(256) void k_softmax(const float* __restrict__ x,
                                                 float* __restrict__ y) {
  int row = blockIdx.x; int t = threadIdx.x;
  __shared__ float red[256];
  float4 v = ((const float4*)(x + (size_t)row * SS))[t];
  red[t] = fmaxf(fmaxf(v.x, v.y), fmaxf(v.z, v.w));
  __syncthreads();
  for (int o = 128; o > 0; o >>= 1) { if (t < o) red[t] = fmaxf(red[t], red[t+o]); __syncthreads(); }
  float m = red[0];
  __syncthreads();
  float4 e = make_float4(__expf(v.x - m), __expf(v.y - m), __expf(v.z - m), __expf(v.w - m));
  red[t] = e.x + e.y + e.z + e.w;
  __syncthreads();
  for (int o = 128; o > 0; o >>= 1) { if (t < o) red[t] += red[t+o]; __syncthreads(); }
  float inv = 1.0f / red[0];
  float4 r = make_float4(e.x * inv, e.y * inv, e.z * inv, e.w * inv);
  ((float4*)(y + (size_t)row * SS))[t] = r;
}

// ---------------------------------------------------------------------------
// 8. attn @ V: recompute energy tiles by WMMA, softmax via stats, P -> LDS
//    (C-layout -> A-layout), WMMA against h_norm^T. Output out_merge (bf16).
// ---------------------------------------------------------------------------
__global__ __launch_bounds__(256) void k_attnv(const bf16_t* __restrict__ Qbf,
                                               const bf16_t* __restrict__ Kbf,
                                               const bf16_t* __restrict__ hnT,
                                               const float2* __restrict__ stats,
                                               bf16_t* __restrict__ OMbf) {
  __shared__ __align__(16) bf16_t psm[8 * 16 * 32];  // 8 KB: per-wave 16x32 P tile
  int bh = blockIdx.x >> 3;
  int qg = blockIdx.x & 7;
  int w  = threadIdx.x >> 5;
  int l  = threadIdx.x & 31;
  int b = bh >> 4, h = bh & 15;
  int qt = qg * 8 + w;
  int q0 = qt * 16;
  bf16_t* ptile = &psm[w * 512];

  v16bf aq0 = load_frag_a(Qbf, DD, b * SS + q0, h * HD);
  v16bf aq1 = load_frag_a(Qbf, DD, b * SS + q0, h * HD + 32);

  int mb = (l < 16) ? 0 : 8;
  int n  = l & 15;
  float rmax[8], rinv[8];
  #pragma unroll
  for (int v = 0; v < 8; ++v) {
    float2 st = stats[(size_t)bh * SS + q0 + mb + v];
    rmax[v] = st.x; rinv[v] = 1.0f / st.y;
  }

  v8f o0 = {}, o1 = {}, o2 = {}, o3 = {};
  for (int kc = 0; kc < 32; ++kc) {
    #pragma unroll
    for (int half = 0; half < 2; ++half) {
      int k0 = kc * 32 + half * 16;
      v8f e = {};
      e = wmma_bf16(aq0, load_frag_bT(Kbf, DD, b * SS + k0, h * HD),      e);
      e = wmma_bf16(aq1, load_frag_bT(Kbf, DD, b * SS + k0, h * HD + 32), e);
      #pragma unroll
      for (int v = 0; v < 8; ++v) {
        float p = __expf(ESCALE * e[v] - rmax[v]) * rinv[v];
        ptile[(mb + v) * 32 + half * 16 + n] = (bf16_t)p;
      }
    }
    // Same-wave LDS RAW: DS ops are in-order per wave; compiler inserts DScnt waits.
    v16bf ap = load_frag_a(ptile, 32, 0, 0);
    int s0 = kc * 32;
    o0 = wmma_bf16(ap, load_frag_bT(hnT, SS, b * DD + h * HD +  0, s0), o0);
    o1 = wmma_bf16(ap, load_frag_bT(hnT, SS, b * DD + h * HD + 16, s0), o1);
    o2 = wmma_bf16(ap, load_frag_bT(hnT, SS, b * DD + h * HD + 32, s0), o2);
    o3 = wmma_bf16(ap, load_frag_bT(hnT, SS, b * DD + h * HD + 48, s0), o3);
  }
  #pragma unroll
  for (int v = 0; v < 8; ++v) {
    size_t row = (size_t)(b * SS + q0 + mb + v);
    bf16_t* out = OMbf + row * DD + h * HD + n;
    out[ 0] = (bf16_t)o0[v];
    out[16] = (bf16_t)o1[v];
    out[32] = (bf16_t)o2[v];
    out[48] = (bf16_t)o3[v];
  }
}

// ---------------------------------------------------------------------------
// 9. y = h + alpha * (out_merge @ o_fwd^T)
// ---------------------------------------------------------------------------
__global__ __launch_bounds__(256) void k_final(const bf16_t* __restrict__ OM,
                                               const bf16_t* __restrict__ Wo,
                                               const float* __restrict__ h,
                                               float* __restrict__ y, float alpha) {
  int wid = (int)((blockIdx.x * blockDim.x + threadIdx.x) >> 5);
  int tm = wid >> 6, tn = wid & 63;
  int m0 = tm * 16, n0 = tn * 16;
  v8f acc = {};
  for (int k0 = 0; k0 < DD; k0 += 32)
    acc = wmma_bf16(load_frag_a(OM, DD, m0, k0), load_frag_bT(Wo, DD, n0, k0), acc);
  int l = threadIdx.x & 31;
  int n  = n0 + (l & 15);
  int mb = m0 + ((l < 16) ? 0 : 8);
  for (int v = 0; v < 8; ++v) {
    size_t idx = (size_t)(mb + v) * DD + n;
    y[idx] = h[idx] + alpha * acc[v];
  }
}

// ---------------------------------------------------------------------------
// Host launch
// ---------------------------------------------------------------------------
extern "C" void kernel_launch(void* const* d_in, const int* in_sizes, int n_in,
                              void* d_out, int out_size, void* d_ws, size_t ws_size,
                              hipStream_t stream) {
  (void)in_sizes; (void)n_in; (void)out_size; (void)ws_size;
  const float* h    = (const float*)d_in[0];
  const float* qf   = (const float*)d_in[1];
  const float* kf   = (const float*)d_in[2];
  const float* of   = (const float*)d_in[3];
  const float* qfc  = (const float*)d_in[4];
  const float* kfc  = (const float*)d_in[5];

  float* out   = (float*)d_out;
  float* y     = out;                         // (B,S,D)
  float* qdiff = out + (size_t)1 * 2097152;   // (B,H,S,hd)
  float* kdiff = out + (size_t)2 * 2097152;
  float* vzero = out + (size_t)3 * 2097152;
  float* ldiff = out + (size_t)4 * 2097152;   // (B,H,S,S)
  float* attm  = out + (size_t)4 * 2097152 + 33554432;  // also logits scratch

  char* ws = (char*)d_ws;
  bf16_t* hnbf = (bf16_t*)(ws + 0);
  bf16_t* hnT  = (bf16_t*)(ws +  4194304);
  bf16_t* wq   = (bf16_t*)(ws +  8388608);
  bf16_t* wk   = (bf16_t*)(ws + 10485760);
  bf16_t* wqc  = (bf16_t*)(ws + 12582912);
  bf16_t* wkc  = (bf16_t*)(ws + 14680064);
  bf16_t* wo   = (bf16_t*)(ws + 16777216);
  bf16_t* Qbf  = (bf16_t*)(ws + 18874368);
  bf16_t* Kbf  = (bf16_t*)(ws + 23068672);
  bf16_t* Qrbf = (bf16_t*)(ws + 27262976);
  bf16_t* Krbf = (bf16_t*)(ws + 31457280);
  bf16_t* OMbf = (bf16_t*)(ws + 35651584);
  float2* stats = (float2*)(ws + 39845888);

  const float alpha = sqrtf(1.0f / (12.0f * 1024.0f));

  k_rmsnorm<<<ROWS, 256, 0, stream>>>(h, hnbf, vzero);
  k_tobf16 <<<1024, 256, 0, stream>>>(qf,  wq);
  k_tobf16 <<<1024, 256, 0, stream>>>(kf,  wk);
  k_tobf16 <<<1024, 256, 0, stream>>>(qfc, wqc);
  k_tobf16 <<<1024, 256, 0, stream>>>(kfc, wkc);
  k_tobf16 <<<1024, 256, 0, stream>>>(of,  wo);
  k_transpose<<<ROWS, 256, 0, stream>>>(hnbf, hnT);

  k_proj<<<1024, 256, 0, stream>>>(hnbf, wq, wqc, Qbf, Qrbf, qdiff);
  k_proj<<<1024, 256, 0, stream>>>(hnbf, wk, wkc, Kbf, Krbf, kdiff);

  k_energy<<<32 * 64 * 8, 256, 0, stream>>>(Qbf, Kbf, Qrbf, Krbf, ldiff, attm);
  k_rowstats<<<32768, 256, 0, stream>>>(attm, stats);   // consume logits scratch
  k_softmax <<<32768, 256, 0, stream>>>(ldiff, attm);   // overwrite with att_minus

  k_attnv<<<32 * 8, 256, 0, stream>>>(Qbf, Kbf, hnT, stats, OMbf);
  k_final<<<1024, 256, 0, stream>>>(OMbf, wo, h, y, alpha);
}
